// RateRNN_19447611916935
// MI455X (gfx1250) — matrix operand order repaired and checked by reading, not compiled, not run on hardware
//
#include <hip/hip_runtime.h>
#include <hip/hip_bf16.h>

// ---------------------------------------------------------------------------
// Rate RNN on MI455X (gfx1250, wave32, WMMA).
//   x_{t} = 0.9*x_{t-1} + sig(x_{t-1}) @ (0.1*w_eff) + u_t @ (0.1*w_in) + n_t
// 8 blocks x 512 threads (16 waves); block owns an independent 16-row batch
// slice for all T=1000 steps.  Each wave owns TWO 16x16 column tiles sharing
// one LDS A-fragment per chunk (2-way WMMA ILP, half the LDS traffic).
// Tile-0 weights: 144 register-resident VGPRs.  Tile-1 weights: streamed
// from L2 every step with a 1-chunk-ahead rotation; an opaque asm touch on
// an SGPR offset (not the pointer!) stops LICM from hoisting/spilling the
// loads while keeping global address-space provenance (global_load, not
// flat_load, so the weight stream doesn't tick DScnt).  Activations live in
// a double-buffered LDS slab (padded rows, conflict-free ds_load_b128).
// Per-step sync: one __syncthreads.
// ---------------------------------------------------------------------------

typedef __attribute__((ext_vector_type(16))) __bf16       v16bf;
typedef __attribute__((ext_vector_type(8)))  float        v8f;
typedef __attribute__((ext_vector_type(4)))  float        v4f;
typedef __attribute__((ext_vector_type(4)))  unsigned int v4u;

#define N_IN    64
#define N_REC   512
#define N_OUT   32
#define BATCH   128
#define T_STEPS 1000
#define KTOT    576            // 512 recurrent + 64 input columns, fused GEMM
#define ALPHA   0.1f           // dt/tau

#define SLAB_STRIDE 1168       // bytes per LDS row: 576*2 + 16 pad (bank-conflict free)

__device__ __forceinline__ v8f wmma_bf16(v16bf a, v16bf b, v8f c) {
  // (neg_a, A, neg_b, B, c_mod, C, reuse_a, reuse_b)
  return __builtin_amdgcn_wmma_f32_16x16x32_bf16(false, a, false, b,
                                                 (short)0, c, false, false);
}

// A-fragment (16x32 bf16, M x K) from an LDS row. Lane holds M = lane&15;
// elements 0..7 -> K = s..s+7, elements 8..15 -> K = s+16..s+23,
// s = (lane>=16)?8:0.  `p` points at row + (k0+s)*2 bytes.
__device__ __forceinline__ v16bf load_a_lds(const unsigned char* p) {
  union { v4u u[2]; v16bf v; } x;
  x.u[0] = *(const v4u*)(p);        // K: s   .. s+7   (16 B)
  x.u[1] = *(const v4u*)(p + 32);   // K: s+16.. s+23  (16 B)
  return x.v;
}

// A-fragment built from an fp32 row (used by the output-projection kernel).
__device__ __forceinline__ v16bf load_a_f32(const float* __restrict__ p) {
  v4f f0 = *(const v4f*)(p);
  v4f f1 = *(const v4f*)(p + 4);
  v4f f2 = *(const v4f*)(p + 16);
  v4f f3 = *(const v4f*)(p + 20);
  v16bf a;
#pragma unroll
  for (int i = 0; i < 4; ++i) {
    a[i]      = (__bf16)f0[i];
    a[4 + i]  = (__bf16)f1[i];
    a[8 + i]  = (__bf16)f2[i];
    a[12 + i] = (__bf16)f3[i];
  }
  return a;
}

// B-fragment (32x16 bf16, K x N): lane holds N = lane&15; elements 0..7 -> K = s..s+7,
// elements 8..15 -> K = s+16..s+23, s = k0 + ((lane>=16)?8:0). `p` pre-offset.
__device__ __forceinline__ v16bf load_b(const __bf16* p) {
  union { v4u u[2]; v16bf v; } x;
  x.u[0] = *(const v4u*)(p);
  x.u[1] = *(const v4u*)(p + 16);
  return x.v;
}

// ---------------------------------------------------------------------------
// Prep: w_eff / w_out_eff / w_in outputs + bf16-swizzled weight matrices.
// ---------------------------------------------------------------------------
__global__ void prep_kernel(const float* __restrict__ w_in,
                            const float* __restrict__ w_rec,
                            const float* __restrict__ w_out,
                            const float* __restrict__ d_sign,
                            float* __restrict__ out_weff,
                            float* __restrict__ out_wouteff,
                            float* __restrict__ out_win,
                            __bf16* __restrict__ Wt,      // [N_REC rows n][KTOT]
                            __bf16* __restrict__ Wto) {   // [N_OUT rows o][N_REC]
  const int idx = blockIdx.x * blockDim.x + threadIdx.x;
  const int stride = gridDim.x * blockDim.x;
  for (int i = idx; i < N_REC * N_REC; i += stride) {      // w_eff + 0.1*w_eff^T
    const int k = i >> 9, n = i & 511;
    const float we = d_sign[k] * fabsf(w_rec[i]);
    out_weff[i] = we;
    Wt[(size_t)n * KTOT + k] = (__bf16)(ALPHA * we);
  }
  for (int i = idx; i < N_IN * N_REC; i += stride) {       // w_in + 0.1*w_in^T
    const int k = i >> 9, n = i & 511;
    const float w = w_in[i];
    out_win[i] = w;
    Wt[(size_t)n * KTOT + N_REC + k] = (__bf16)(ALPHA * w);
  }
  for (int i = idx; i < N_REC * N_OUT; i += stride) {      // w_out_eff (+ ^T)
    const int k = i >> 5, o = i & 31;
    const float we = d_sign[k] * fabsf(w_out[i]);
    out_wouteff[i] = we;
    Wto[(size_t)o * N_REC + k] = (__bf16)we;
  }
}

// ---------------------------------------------------------------------------
// Persistent recurrence kernel: 8 blocks x 512 threads (16 waves).
// ---------------------------------------------------------------------------
__global__ void __launch_bounds__(512)
rnn_kernel(const float* __restrict__ inputs,   // (B,T,N_IN)
           const float* __restrict__ noise,    // (B,T,N_REC)
           const __bf16* __restrict__ Wt,      // [N_REC][KTOT]
           float* __restrict__ x_rec,          // (B,T,N_REC)
           float* __restrict__ r_rec) {        // (B,T,N_REC)
  __shared__ __align__(16) unsigned char slab[2][16 * SLAB_STRIDE];

  const int tid   = threadIdx.x;
  const int lane  = tid & 31;
  const int wv    = tid >> 5;         // 0..15
  const int b0    = blockIdx.x * 16;  // batch slice
  const int n0    = wv * 16;          // tile 0 columns
  const int n1    = 256 + wv * 16;    // tile 1 columns
  const int laneHi = (lane >= 16);
  const int mRow = lane & 15;         // A: M row; B: N col; C/D: N col
  const int sA = laneHi ? 8 : 0;
  const int sB = laneHi ? 8 : 0;
  const int cM = laneHi ? 8 : 0;      // C/D row offset

  // Tile-0 weights register-resident (18 fragments = 144 VGPRs).
  const __bf16* wrow0 = Wt + (size_t)(n0 + mRow) * KTOT + sB;
  v16bf Wb[18];
#pragma unroll
  for (int c = 0; c < 18; ++c) Wb[c] = load_b(wrow0 + c * 32);
  // Tile-1 weights streamed from L2 each step.
  const __bf16* wrow1 = Wt + (size_t)(n1 + mRow) * KTOT + sB;

  // ---- Stage step 0 into slab[0]: r = sigmoid(0) = 0.5, u = inputs[:,0,:].
  for (int i = tid; i < 16 * 64; i += 512) {               // 512 bf16 cols / row
    const int row = i >> 6, c8 = i & 63;
    v4u h; h.x = h.y = h.z = h.w = 0x3F003F00u;            // bf16 0.5 pair
    *(v4u*)(slab[0] + row * SLAB_STRIDE + c8 * 16) = h;
  }
  if (tid < 256) {                                         // u_0 -> bf16 tail cols
    const int row = tid >> 4, col = (tid & 15) * 4;
    v4f f = *(const v4f*)(inputs + ((size_t)(b0 + row) * T_STEPS + 0) * N_IN + col);
    union { __bf16 h[4]; unsigned long long q; } pk;
#pragma unroll
    for (int j = 0; j < 4; ++j) pk.h[j] = (__bf16)f[j];
    *(unsigned long long*)(slab[0] + row * SLAB_STRIDE + 1024 + col * 2) = pk.q;
  }
  __syncthreads();

  v8f x0, x1;
#pragma unroll
  for (int r = 0; r < 8; ++r) { x0[r] = 0.0f; x1[r] = 0.0f; }

  for (int t = 0; t < T_STEPS; ++t) {
    const unsigned char* cur = slab[t & 1];
    unsigned char*       nxt = slab[(t + 1) & 1];

    // Opaque per-iteration SGPR offset: defeats LICM of the 18 streamed
    // weight loads (round-4 spill failure) while preserving the global
    // address-space provenance of the base pointer (round-5 flat_load /
    // DScnt-coupling failure).  Costs zero instructions.
    unsigned woff = 0;
    asm volatile("" : "+s"(woff));
    const __bf16* w1 = wrow1 + woff;

    // Prefetch this step's noise (consumed only after the WMMA chains).
    float nz0[8], nz1[8];
#pragma unroll
    for (int r = 0; r < 8; ++r) {
      const size_t row = ((size_t)(b0 + r + cM) * T_STEPS + t) * N_REC + mRow;
      nz0[r] = noise[row + n0];
      nz1[r] = noise[row + n1];
    }

    // Stage u_{t+1} into the write-side slab (overlaps the WMMA chains).
    if (t + 1 < T_STEPS && tid < 256) {
      const int row = tid >> 4, col = (tid & 15) * 4;
      v4f f = *(const v4f*)(inputs + ((size_t)(b0 + row) * T_STEPS + (t + 1)) * N_IN + col);
      union { __bf16 h[4]; unsigned long long q; } pk;
#pragma unroll
      for (int j = 0; j < 4; ++j) pk.h[j] = (__bf16)f[j];
      *(unsigned long long*)(nxt + row * SLAB_STRIDE + 1024 + col * 2) = pk.q;
    }

    // Two accumulator chains seeded with the leak term.
    v8f acc0, acc1;
#pragma unroll
    for (int r = 0; r < 8; ++r) {
      acc0[r] = (1.0f - ALPHA) * x0[r];
      acc1[r] = (1.0f - ALPHA) * x1[r];
    }

    // 18 K-chunks: shared A from LDS; tile-0 B from VGPRs, tile-1 B from L2
    // with a 1-chunk-ahead register rotation (partial loadcnt waits).
    const unsigned char* arow = cur + mRow * SLAB_STRIDE + sA * 2;
    v16bf b1 = load_b(w1);
#pragma unroll
    for (int c = 0; c < 18; ++c) {
      v16bf b1n = b1;
      if (c < 17) b1n = load_b(w1 + (c + 1) * 32);
      const v16bf a = load_a_lds(arow + c * 64);
      acc0 = wmma_bf16(a, Wb[c], acc0);
      acc1 = wmma_bf16(a, b1, acc1);
      b1 = b1n;
    }

#pragma unroll
    for (int r = 0; r < 8; ++r) { acc0[r] += nz0[r]; acc1[r] += nz1[r]; }
    x0 = acc0; x1 = acc1;

    // Epilogue: x_t / r_t to HBM; bf16 r_t into the write-side LDS slab.
#pragma unroll
    for (int r = 0; r < 8; ++r) {
      const size_t row = ((size_t)(b0 + r + cM) * T_STEPS + t) * N_REC + mRow;
      const float xv0 = x0[r], xv1 = x1[r];
      const float rv0 = 1.0f / (1.0f + __expf(-xv0));
      const float rv1 = 1.0f / (1.0f + __expf(-xv1));
      x_rec[row + n0] = xv0;
      x_rec[row + n1] = xv1;
      r_rec[row + n0] = rv0;
      r_rec[row + n1] = rv1;
      unsigned char* srow = nxt + (r + cM) * SLAB_STRIDE + mRow * 2;
      *(__bf16*)(srow + n0 * 2) = (__bf16)rv0;
      *(__bf16*)(srow + n1 * 2) = (__bf16)rv1;
    }
    __syncthreads();
  }
}

// ---------------------------------------------------------------------------
// o_rec = r_rec(B*T, 512) @ w_out_eff(512, 32): 8000 x 2 tiles, 1 per wave.
// ---------------------------------------------------------------------------
__global__ void __launch_bounds__(256)
out_proj_kernel(const float* __restrict__ r_rec,   // (B*T, N_REC)
                const __bf16* __restrict__ Wto,    // [N_OUT][N_REC]
                float* __restrict__ o_rec) {       // (B*T, N_OUT)
  const int lane  = threadIdx.x & 31;
  const int gwave = blockIdx.x * 8 + (threadIdx.x >> 5);  // 0..15999
  const int mt = gwave >> 1;          // 0..7999
  const int nt = gwave & 1;           // 0..1
  const int m0 = mt * 16, o0 = nt * 16;
  const int laneHi = (lane >= 16);
  const int mRow = lane & 15;
  const int sA = laneHi ? 8 : 0;
  const int sB = laneHi ? 8 : 0;
  const int cM = laneHi ? 8 : 0;

  const float* abase = r_rec + (size_t)(m0 + mRow) * N_REC + sA;
  const __bf16* bbase = Wto + (size_t)(o0 + mRow) * N_REC + sB;

  v8f acc;
#pragma unroll
  for (int r = 0; r < 8; ++r) acc[r] = 0.0f;
#pragma unroll
  for (int c = 0; c < 16; ++c)
    acc = wmma_bf16(load_a_f32(abase + c * 32), load_b(bbase + c * 32), acc);
#pragma unroll
  for (int r = 0; r < 8; ++r)
    o_rec[(size_t)(m0 + r + cM) * N_OUT + o0 + mRow] = acc[r];
}

// ---------------------------------------------------------------------------
extern "C" void kernel_launch(void* const* d_in, const int* in_sizes, int n_in,
                              void* d_out, int out_size, void* d_ws, size_t ws_size,
                              hipStream_t stream) {
  const float* inputs = (const float*)d_in[0];   // (128,1000,64)
  const float* noise  = (const float*)d_in[1];   // (128,1000,512)
  const float* w_in   = (const float*)d_in[2];   // (64,512)
  const float* w_rec  = (const float*)d_in[3];   // (512,512)
  const float* w_out  = (const float*)d_in[4];   // (512,32)
  const float* d_sign = (const float*)d_in[5];   // (512,)

  float* out = (float*)d_out;
  float* x_rec     = out;
  float* r_rec     = x_rec + (size_t)BATCH * T_STEPS * N_REC;
  float* o_rec     = r_rec + (size_t)BATCH * T_STEPS * N_REC;
  float* o_weff    = o_rec + (size_t)BATCH * T_STEPS * N_OUT;
  float* o_wouteff = o_weff + (size_t)N_REC * N_REC;
  float* o_win     = o_wouteff + (size_t)N_REC * N_OUT;

  char* ws = (char*)d_ws;
  __bf16* Wt  = (__bf16*)(ws);                   // 512*576*2 = 589824 B
  __bf16* Wto = (__bf16*)(ws + 589824);          // 32*512*2  =  32768 B

  prep_kernel<<<256, 256, 0, stream>>>(w_in, w_rec, w_out, d_sign,
                                       o_weff, o_wouteff, o_win, Wt, Wto);
  rnn_kernel<<<BATCH / 16, 512, 0, stream>>>(inputs, noise, Wt, x_rec, r_rec);
  out_proj_kernel<<<(BATCH * T_STEPS / 16) * 2 / 8, 256, 0, stream>>>(r_rec, Wto, o_rec);
}
